// GraphMoEHierarchical_70875550319089
// MI455X (gfx1250) — matrix operand order repaired and checked by reading, not compiled, not run on hardware
//
#include <hip/hip_runtime.h>
#include <math.h>

typedef __attribute__((ext_vector_type(16))) __bf16 v16bf;
typedef __attribute__((ext_vector_type(8)))  float  v8f;

#define Hdim 128
#define INdim 6
#define Gnum 64
#define NGc 2
#define EPGc 2
#define Lc 3
#define NMAT (NGc*EPGc*Lc)   // 12 weight matrices [128x128]

// ---------------------------------------------------------------- utilities
__global__ void fill_f32(float* __restrict__ p, float v, int n) {
  int i = blockIdx.x * blockDim.x + threadIdx.x;
  if (i < n) p[i] = v;
}

__global__ void fill4_f32(float4* __restrict__ p, float v, int n4) {
  int i = blockIdx.x * blockDim.x + threadIdx.x;
  if (i < n4) p[i] = make_float4(v, v, v, v);
}

// ------------------------------------------------- encoder: h = relu(x@W+b)
__global__ void encoder_kernel(const float* __restrict__ x, const float* __restrict__ W,
                               const float* __restrict__ b, float* __restrict__ h, int N) {
  int tid = blockIdx.x * blockDim.x + threadIdx.x;
  int n = tid >> 5;
  int c = (tid & 31) << 2;
  if (n >= N) return;
  float xv[INdim];
#pragma unroll
  for (int k = 0; k < INdim; ++k) xv[k] = x[(size_t)n * INdim + k];
  float4 acc = *(const float4*)(b + c);
#pragma unroll
  for (int k = 0; k < INdim; ++k) {
    float4 w = *(const float4*)(W + k * Hdim + c);
    acc.x += xv[k] * w.x; acc.y += xv[k] * w.y;
    acc.z += xv[k] * w.z; acc.w += xv[k] * w.w;
  }
  acc.x = fmaxf(acc.x, 0.f); acc.y = fmaxf(acc.y, 0.f);
  acc.z = fmaxf(acc.z, 0.f); acc.w = fmaxf(acc.w, 0.f);
  *(float4*)(h + (size_t)n * Hdim + c) = acc;
}

// --------------------------------- per-graph scalar stats (segment sums)
__global__ void node_stats_kernel(const float* __restrict__ x, const int* __restrict__ batch,
                                  float* __restrict__ cnt, float* __restrict__ sx,
                                  float* __restrict__ sx2, int N) {
  int n = blockIdx.x * blockDim.x + threadIdx.x;
  if (n >= N) return;
  int b = batch[n];
  float s = 0.f, s2 = 0.f;
#pragma unroll
  for (int k = 0; k < INdim; ++k) {
    float v = x[(size_t)n * INdim + k];
    s += v; s2 += v * v;
  }
  atomicAdd(&cnt[b], 1.f);
  atomicAdd(&sx[b], s);
  atomicAdd(&sx2[b], s2);
}

// ----------------------- per-edge: degree(dst) and edge-count per graph(src)
__global__ void edge_stats_kernel(const int* __restrict__ src, const int* __restrict__ dst,
                                  const int* __restrict__ batch,
                                  float* __restrict__ degf, float* __restrict__ ecnt, int E) {
  int e = blockIdx.x * blockDim.x + threadIdx.x;
  if (e >= E) return;
  atomicAdd(&degf[dst[e]], 1.f);
  atomicAdd(&ecnt[batch[src[e]]], 1.f);
}

__global__ void invdeg_kernel(const float* __restrict__ degf, float* __restrict__ invdeg, int N) {
  int n = blockIdx.x * blockDim.x + threadIdx.x;
  if (n >= N) return;
  invdeg[n] = 1.f / fmaxf(degf[n], 1.f);
}

// --------------------------- graph-level router (G=64, one block, tiny)
__global__ void graph_router_kernel(const float* __restrict__ cnt, const float* __restrict__ sx,
                                    const float* __restrict__ sx2,
                                    const float* __restrict__ grW1, const float* __restrict__ grb1,
                                    const float* __restrict__ grW2, const float* __restrict__ grb2,
                                    float* __restrict__ gprobs) {
  int g = threadIdx.x;
  if (g >= Gnum) return;
  float tot = fmaxf(cnt[g] * (float)INdim, 1.f);
  float mean = sx[g] / tot;
  float var = fmaxf(sx2[g] / tot - mean * mean, 0.f);
  float sd = sqrtf(var);
  float l0 = grb2[0], l1 = grb2[1];
  for (int j = 0; j < Hdim; ++j) {
    float hj = fmaxf(mean * grW1[j] + sd * grW1[Hdim + j] + grb1[j], 0.f);
    l0 += hj * grW2[j * NGc + 0];
    l1 += hj * grW2[j * NGc + 1];
  }
  float mx = fmaxf(l0, l1);
  float e0 = expf(l0 - mx), e1 = expf(l1 - mx);
  float inv = 1.f / (e0 + e1);
  gprobs[g * NGc + 0] = e0 * inv;
  gprobs[g * NGc + 1] = e1 * inv;
}

// ------------------------------------- node router -> gate[n][g*EPG+e]
__global__ __launch_bounds__(128)
void node_router_kernel(const float* __restrict__ h, const int* __restrict__ batch,
                        const float* __restrict__ cnt, const float* __restrict__ ecnt,
                        const float* __restrict__ gprobs,
                        const float* __restrict__ nrW1, const float* __restrict__ nrb1,
                        const float* __restrict__ nrW2, const float* __restrict__ nrb2,
                        float* __restrict__ gate, int N) {
  __shared__ float hs[16][Hdim];
  __shared__ float hid[16][Hdim];
  int j = threadIdx.x;
  int row0 = blockIdx.x * 16;
#pragma unroll
  for (int r = 0; r < 16; ++r) {
    int n = row0 + r;
    hs[r][j] = (n < N) ? h[(size_t)n * Hdim + j] : 0.f;
  }
  __syncthreads();
  float acc[16];
  float bj = nrb1[j];
#pragma unroll
  for (int r = 0; r < 16; ++r) acc[r] = bj;
  for (int k = 0; k < Hdim; ++k) {
    float w = nrW1[k * Hdim + j];
#pragma unroll
    for (int r = 0; r < 16; ++r) acc[r] += hs[r][k] * w;
  }
  float w128 = nrW1[Hdim * Hdim + j];
  float w129 = nrW1[(Hdim + 1) * Hdim + j];
#pragma unroll
  for (int r = 0; r < 16; ++r) {
    int n = row0 + r;
    if (n < N) {
      int b = batch[n];
      float sf0 = logf(cnt[b]);
      float sf1 = log1pf(ecnt[b]);
      acc[r] += sf0 * w128 + sf1 * w129;
    }
    hid[r][j] = fmaxf(acc[r], 0.f);
  }
  __syncthreads();
  if (j < 16) {
    int n = row0 + j;
    if (n < N) {
      float l0 = nrb2[0], l1 = nrb2[1];
      for (int t = 0; t < Hdim; ++t) {
        float hv = hid[j][t];
        l0 += hv * nrW2[t * EPGc + 0];
        l1 += hv * nrW2[t * EPGc + 1];
      }
      float mx = fmaxf(l0, l1);
      float e0 = expf(l0 - mx), e1 = expf(l1 - mx);
      float inv = 1.f / (e0 + e1);
      float p0 = e0 * inv, p1 = e1 * inv;
      int am = (p1 > p0) ? 1 : 0;            // jnp.argmax: ties -> index 0
      float pw = am ? p1 : p0;
      int b = batch[n];
      float g0 = gprobs[b * NGc + 0], g1 = gprobs[b * NGc + 1];
      float s0 = (am == 0) ? pw : 0.f;
      float s1 = (am == 1) ? pw : 0.f;
      gate[(size_t)n * 4 + 0] = s0 * g0;     // g=0,e=0
      gate[(size_t)n * 4 + 1] = s1 * g0;     // g=0,e=1
      gate[(size_t)n * 4 + 2] = s0 * g1;     // g=1,e=0
      gate[(size_t)n * 4 + 3] = s1 * g1;     // g=1,e=1
    }
  }
}

// ----------- pre-swizzle expert weights fp32 -> bf16 in WMMA B-layout
// Wb[mat][kb(4)][nb(8)][lane(32)][i(16)]: col = nb*16+(lane&15),
// K = kb*32 + (lane>=16 ? 16 : 0) + i   (contiguous k order, same as A; 16-bit B 32x16)
__global__ void swizzle_w_kernel(const float* __restrict__ expW, __bf16* __restrict__ Wb) {
  int tid = blockIdx.x * blockDim.x + threadIdx.x;
  int i = tid & 15;
  int lane = (tid >> 4) & 31;
  int nb = (tid >> 9) & 7;
  int kb = (tid >> 12) & 3;
  int mat = tid >> 14;
  if (mat >= NMAT) return;
  int col = nb * 16 + (lane & 15);
  int K = kb * 32 + ((lane >> 4) << 4) + i;
  float v = expW[((size_t)mat * Hdim + K) * Hdim + col];
  Wb[tid] = (__bf16)v;
}

// ------------------- scatter aggregation: m[dst] += he[src], 4 cols/thread
__global__ void agg_kernel(const float* __restrict__ he, const int* __restrict__ src,
                           const int* __restrict__ dst, float* __restrict__ m, int E) {
  int tid = blockIdx.x * blockDim.x + threadIdx.x;
  int e = tid >> 5;
  if (e >= E) return;
  int c4 = (tid & 31) << 2;
  int s = src[e], d = dst[e];
  float4 v = *(const float4*)(he + (size_t)s * Hdim + c4);
  float* mp = m + (size_t)d * Hdim + c4;
  atomicAdd(mp + 0, v.x);
  atomicAdd(mp + 1, v.y);
  atomicAdd(mp + 2, v.z);
  atomicAdd(mp + 3, v.w);
}

// ------------------- expert layer GEMM: out = act((he + m*invdeg) @ W + b)
// One wave -> 16 rows x 128 cols; all 32 v_wmma_f32_16x16x32_bf16 issued
// before the (uniform-branch) epilogue so the 8 independent accumulation
// chains can interleave and hide the WMMA->VALU D-read hazard.
// relu_mode=1: out = relu(z). relu_mode=0: out += z * gate[row*4+gidx].
__global__ __launch_bounds__(128)
void expert_gemm_kernel(const float* __restrict__ he, const float* __restrict__ m,
                        const float* __restrict__ invdeg,
                        const __bf16* __restrict__ WbMat, const float* __restrict__ bias,
                        float* __restrict__ out, const float* __restrict__ gate,
                        int gidx, int relu_mode, int N) {
  int wave = threadIdx.x >> 5;
  int lane = threadIdx.x & 31;
  int row0 = (blockIdx.x * 4 + wave) * 16;
  if (row0 >= N) return;                 // wave-uniform exit
  // keep the 32KB weight matrix hot (gfx1250 global_prefetch_b8)
  __builtin_prefetch(WbMat + (size_t)threadIdx.x * 128, 0, 1);

  int lo = lane & 15;
  int hi = lane >> 4;                    // 0 or 1
  // force the tile-full predicate scalar so the epilogue is one s_cbranch
  int full = __builtin_amdgcn_readfirstlane((row0 + 16 <= N) ? 1 : 0);
  int arow = row0 + lo;
  if (arow > N - 1) arow = N - 1;        // clamp: A row m only affects D row m,
                                         // overflow rows are never stored
  float id = invdeg[arow];
  const float* hrow = he + (size_t)arow * Hdim;
  const float* mrow = m + (size_t)arow * Hdim;

  // A fragments: 16-bit A 16x32, contiguous k = kb*32 + 16*hi + i (same as Wb), branch-free vector loads
  v16bf a[4];
#pragma unroll
  for (int kb = 0; kb < 4; ++kb) {
    int k0 = kb * 32 + hi * 16;
    float4 h0 = *(const float4*)(hrow + k0);
    float4 h1 = *(const float4*)(hrow + k0 + 4);
    float4 h2 = *(const float4*)(hrow + k0 + 8);
    float4 h3 = *(const float4*)(hrow + k0 + 12);
    float4 q0 = *(const float4*)(mrow + k0);
    float4 q1 = *(const float4*)(mrow + k0 + 4);
    float4 q2 = *(const float4*)(mrow + k0 + 8);
    float4 q3 = *(const float4*)(mrow + k0 + 12);
    a[kb][0]  = (__bf16)(h0.x + q0.x * id);
    a[kb][1]  = (__bf16)(h0.y + q0.y * id);
    a[kb][2]  = (__bf16)(h0.z + q0.z * id);
    a[kb][3]  = (__bf16)(h0.w + q0.w * id);
    a[kb][4]  = (__bf16)(h1.x + q1.x * id);
    a[kb][5]  = (__bf16)(h1.y + q1.y * id);
    a[kb][6]  = (__bf16)(h1.z + q1.z * id);
    a[kb][7]  = (__bf16)(h1.w + q1.w * id);
    a[kb][8]  = (__bf16)(h2.x + q2.x * id);
    a[kb][9]  = (__bf16)(h2.y + q2.y * id);
    a[kb][10] = (__bf16)(h2.z + q2.z * id);
    a[kb][11] = (__bf16)(h2.w + q2.w * id);
    a[kb][12] = (__bf16)(h3.x + q3.x * id);
    a[kb][13] = (__bf16)(h3.y + q3.y * id);
    a[kb][14] = (__bf16)(h3.z + q3.z * id);
    a[kb][15] = (__bf16)(h3.w + q3.w * id);
  }

  // all 32 WMMAs up front: 8 independent accumulation chains
  v8f c[8];
#pragma unroll
  for (int nb = 0; nb < 8; ++nb) {
    c[nb] = (v8f){};
#pragma unroll
    for (int kb = 0; kb < 4; ++kb) {
      v16bf b = *(const v16bf*)(WbMat + (((size_t)kb * 8 + nb) * 32 + lane) * 16);
      c[nb] = __builtin_amdgcn_wmma_f32_16x16x32_bf16(false, a[kb], false, b,
                                                      (short)0, c[nb], false, false);
    }
  }

  // epilogue (one scalar branch on full/relu_mode)
  int rbase = row0 + hi * 8;             // C/D layout: VGPR j -> M = j + 8*hi
  if (full) {
    if (relu_mode) {
#pragma unroll
      for (int nb = 0; nb < 8; ++nb) {
        int col = nb * 16 + lo;
        float bv = bias[col];
        float* orow = out + (size_t)rbase * Hdim + col;
#pragma unroll
        for (int j = 0; j < 8; ++j)
          orow[(size_t)j * Hdim] = fmaxf(c[nb][j] + bv, 0.f);
      }
    } else {
      float gv[8];
      const float* grow = gate + (size_t)rbase * 4 + gidx;
#pragma unroll
      for (int j = 0; j < 8; ++j) gv[j] = grow[(size_t)j * 4];
#pragma unroll
      for (int nb = 0; nb < 8; ++nb) {
        int col = nb * 16 + lo;
        float bv = bias[col];
        float* orow = out + (size_t)rbase * Hdim + col;
#pragma unroll
        for (int j = 0; j < 8; ++j)
          orow[(size_t)j * Hdim] += (c[nb][j] + bv) * gv[j];
      }
    }
  } else {
#pragma unroll
    for (int nb = 0; nb < 8; ++nb) {
      int col = nb * 16 + lo;
      float bv = bias[col];
      float* orow = out + (size_t)rbase * Hdim + col;
#pragma unroll
      for (int j = 0; j < 8; ++j) {
        int row = rbase + j;
        if (row < N) {
          float z = c[nb][j] + bv;
          if (relu_mode) orow[(size_t)j * Hdim] = fmaxf(z, 0.f);
          else           orow[(size_t)j * Hdim] += z * gate[(size_t)row * 4 + gidx];
        }
      }
    }
  }
}

// ================================================================ launcher
extern "C" void kernel_launch(void* const* d_in, const int* in_sizes, int n_in,
                              void* d_out, int out_size, void* d_ws, size_t ws_size,
                              hipStream_t stream) {
  const float* x      = (const float*)d_in[0];
  const int*   eidx   = (const int*)d_in[1];
  const int*   batch  = (const int*)d_in[2];
  const float* W_enc  = (const float*)d_in[3];
  const float* b_enc  = (const float*)d_in[4];
  const float* grW1   = (const float*)d_in[5];
  const float* grb1   = (const float*)d_in[6];
  const float* grW2   = (const float*)d_in[7];
  const float* grb2   = (const float*)d_in[8];
  const float* nrW1   = (const float*)d_in[9];
  const float* nrb1   = (const float*)d_in[10];
  const float* nrW2   = (const float*)d_in[11];
  const float* nrb2   = (const float*)d_in[12];
  const float* expW   = (const float*)d_in[13];
  const float* expb   = (const float*)d_in[14];

  const int N = in_sizes[0] / INdim;
  const int E = in_sizes[1] / 2;
  const int* src = eidx;
  const int* dst = eidx + E;
  float* out = (float*)d_out;

  // ---- workspace carve-out (256B aligned)
  char* base = (char*)d_ws;
  size_t off = 0;
  auto alloc = [&](size_t bytes) -> char* {
    char* p = base + off;
    off = (off + bytes + 255) & ~(size_t)255;
    return p;
  };
  const size_t NH = (size_t)N * Hdim;
  float* h      = (float*)alloc(NH * 4);
  float* m0     = (float*)alloc(NH * 4);
  float* mbuf   = (float*)alloc(NH * 4);
  float* heA    = (float*)alloc(NH * 4);
  float* heB    = (float*)alloc(NH * 4);
  float* degf   = (float*)alloc((size_t)N * 4);
  float* invdeg = (float*)alloc((size_t)N * 4);
  float* gate   = (float*)alloc((size_t)N * 4 * 4);
  float* cnt    = (float*)alloc(Gnum * 4);
  float* sx     = (float*)alloc(Gnum * 4);
  float* sx2    = (float*)alloc(Gnum * 4);
  float* ecnt   = (float*)alloc(Gnum * 4);
  float* gprobs = (float*)alloc(Gnum * NGc * 4);
  __bf16* Wb    = (__bf16*)alloc((size_t)NMAT * Hdim * Hdim * 2);
  (void)ws_size; (void)n_in;

  auto blk = [](long long total, int b) { return (int)((total + b - 1) / b); };
  const int nh4 = (int)(NH / 4);         // NH = N*128, always /4

  // ---- zero accumulators (every call: deterministic)
  fill4_f32<<<blk(nh4, 256), 256, 0, stream>>>((float4*)m0, 0.f, nh4);
  fill_f32<<<blk(N, 256), 256, 0, stream>>>(degf, 0.f, N);
  fill_f32<<<blk(4 * Gnum, 256), 256, 0, stream>>>(cnt, 0.f, 4 * Gnum); // cnt,sx,sx2,ecnt contiguous
  fill4_f32<<<blk(out_size / 4, 256), 256, 0, stream>>>((float4*)out, 0.f, out_size / 4);

  // ---- encoder + stats + degrees
  encoder_kernel<<<blk((long long)N * 32, 256), 256, 0, stream>>>(x, W_enc, b_enc, h, N);
  node_stats_kernel<<<blk(N, 256), 256, 0, stream>>>(x, batch, cnt, sx, sx2, N);
  edge_stats_kernel<<<blk(E, 256), 256, 0, stream>>>(src, dst, batch, degf, ecnt, E);
  invdeg_kernel<<<blk(N, 256), 256, 0, stream>>>(degf, invdeg, N);

  // ---- routers
  graph_router_kernel<<<1, 64, 0, stream>>>(cnt, sx, sx2, grW1, grb1, grW2, grb2, gprobs);
  node_router_kernel<<<blk(N, 16), 128, 0, stream>>>(h, batch, cnt, ecnt, gprobs,
                                                     nrW1, nrb1, nrW2, nrb2, gate, N);

  // ---- shared layer-0 aggregation of h (reused by all 4 experts)
  agg_kernel<<<blk((long long)E * 32, 256), 256, 0, stream>>>(h, src, dst, m0, E);

  // ---- weight pre-swizzle to bf16 WMMA B-layout
  swizzle_w_kernel<<<blk((long long)NMAT * Hdim * Hdim, 256), 256, 0, stream>>>(expW, Wb);

  // ---- expert pipeline
  const int gemm_grid = blk((long long)N, 64);   // 4 waves x 16 rows per block
  for (int g = 0; g < NGc; ++g) {
    for (int e = 0; e < EPGc; ++e) {
      int matBase = (g * EPGc + e) * Lc;
      int gidx = g * EPGc + e;
      const __bf16* W0 = Wb + (size_t)(matBase + 0) * Hdim * Hdim;
      const __bf16* W1 = Wb + (size_t)(matBase + 1) * Hdim * Hdim;
      const __bf16* W2 = Wb + (size_t)(matBase + 2) * Hdim * Hdim;
      const float* B0 = expb + (size_t)(matBase + 0) * Hdim;
      const float* B1 = expb + (size_t)(matBase + 1) * Hdim;
      const float* B2 = expb + (size_t)(matBase + 2) * Hdim;

      // layer 0: uses shared m0
      expert_gemm_kernel<<<gemm_grid, 128, 0, stream>>>(h, m0, invdeg, W0, B0,
                                                        heA, gate, gidx, 1, N);
      // layer 1
      fill4_f32<<<blk(nh4, 256), 256, 0, stream>>>((float4*)mbuf, 0.f, nh4);
      agg_kernel<<<blk((long long)E * 32, 256), 256, 0, stream>>>(heA, src, dst, mbuf, E);
      expert_gemm_kernel<<<gemm_grid, 128, 0, stream>>>(heA, mbuf, invdeg, W1, B1,
                                                        heB, gate, gidx, 1, N);
      // layer 2 (no relu, gated accumulate into out)
      fill4_f32<<<blk(nh4, 256), 256, 0, stream>>>((float4*)mbuf, 0.f, nh4);
      agg_kernel<<<blk((long long)E * 32, 256), 256, 0, stream>>>(heB, src, dst, mbuf, E);
      expert_gemm_kernel<<<gemm_grid, 128, 0, stream>>>(heB, mbuf, invdeg, W2, B2,
                                                        out, gate, gidx, 0, N);
    }
  }
}
